// ScaledDotProductAttention_22694607192524
// MI455X (gfx1250) — matrix operand order repaired and checked by reading, not compiled, not run on hardware
//
#include <hip/hip_runtime.h>

// ---------------------------------------------------------------------------
// Scaled dot-product attention (forward, cca=0) for MI455X / gfx1250.
// B=32, S=2048, H=64, fp32 in/out. Outputs: ctx [B,S,H] then attn [B,S,S].
// Memory-bound on the 512MB attn write -> recompute QK (cheap WMMA f16)
// instead of spilling unnormalized scores; online softmax in phase 1.
// ---------------------------------------------------------------------------

typedef _Float16 v16h __attribute__((ext_vector_type(16)));
typedef float    v8f  __attribute__((ext_vector_type(8)));

#define WAVES_PER_WG 4
constexpr int   B_ = 32, S_ = 2048, H_ = 64;
constexpr float SCALE_INV = 0.125f;     // 1/sqrt(H) = 1/8, folded into Q
constexpr float SENT      = -60000.0f;  // mask sentinel, representable in f16

// A-matrix (16x32 f16) K pattern: lanes 0-15 cover K {0..7,16..23}, lanes
// 16-31 cover K {8..15,24..31}; VGPR j holds a consecutive K pair.
__device__ __forceinline__ int kpatA(int j, int lane) {
  return ((j < 4) ? 2 * j : 16 + 2 * (j - 4)) + 8 * (lane >> 4);
}
// B-matrix (32x16 f16) K pattern: lanes 0-15 cover K {0..7,16..23}, lanes
// 16-31 cover K {8..15,24..31}; VGPR j holds a consecutive K pair.
__device__ __forceinline__ int kpatB(int j, int lane) {
  return ((j < 4) ? 2 * j : 16 + 2 * (j - 4)) + 8 * (lane >> 4);
}

__device__ __forceinline__ v8f wmma16(v16h a, v16h b, v8f c) {
  // (neg_a, A, neg_b, B, c_mod, C, reuse_a, reuse_b)
  return __builtin_amdgcn_wmma_f32_16x16x32_f16(false, a, false, b,
                                                (short)0, c, false, false);
}

// Load 16 halves where consecutive K pairs are contiguous floats in memory
// (Q rows and K rows: K dim == head dim, stride 1).
__device__ __forceinline__ v16h load_pairs(const float* __restrict__ rowp,
                                           int lane, float scale, bool isA) {
  v16h r;
#pragma unroll
  for (int j = 0; j < 8; ++j) {
    int kk = isA ? kpatA(j, lane) : kpatB(j, lane);
    float2 f = *reinterpret_cast<const float2*>(rowp + kk);  // 8B aligned
    r[2 * j]     = (_Float16)(f.x * scale);
    r[2 * j + 1] = (_Float16)(f.y * scale);
  }
  return r;
}

// V B-tile: K dim is the key dim -> consecutive K pair is stride-H in memory.
__device__ __forceinline__ v16h load_vtile(const float* __restrict__ basep,
                                           int lane) {
  v16h r;
#pragma unroll
  for (int j = 0; j < 8; ++j) {
    int kk = kpatB(j, lane);
    r[2 * j]     = (_Float16)basep[(size_t)kk * H_];
    r[2 * j + 1] = (_Float16)basep[(size_t)(kk + 1) * H_];
  }
  return r;
}

__global__ __launch_bounds__(WAVES_PER_WG * 32)
void sdpa_fwd_kernel(const float* __restrict__ q, const float* __restrict__ k,
                     const float* __restrict__ v,
                     const unsigned char* __restrict__ mask,
                     float* __restrict__ ctx, float* __restrict__ attn) {
  __shared__ alignas(16) _Float16 ptile[WAVES_PER_WG][16][32];

  const int lane = threadIdx.x & 31;
  const int w    = threadIdx.x >> 5;
  const int wid  = blockIdx.x * WAVES_PER_WG + w;  // global wave id
  const int b    = wid >> 7;                       // 128 query tiles / batch
  const int q0   = (wid & 127) * 16;
  const int ncol = lane & 15;                      // N / A-row for this lane
  const int mhalf = (lane >> 4) * 8;               // C-layout row base

  // ---- Q tile -> A regs (scale folded in) ----
  const float* qrow = q + ((size_t)(b * S_) + q0 + ncol) * H_;
  const v16h aq0 = load_pairs(qrow,      lane, SCALE_INV, true);
  const v16h aq1 = load_pairs(qrow + 32, lane, SCALE_INV, true);

  const unsigned char* mrow = mask + ((size_t)(b * S_) + q0) * S_;

  // ---- Phase 1: scores + mask + online softmax (per C-layout slot) ----
  float rm[8], rs[8];
#pragma unroll
  for (int r = 0; r < 8; ++r) { rm[r] = -3.0e38f; rs[r] = 0.0f; }

  for (int nt = 0; nt < S_ / 16; ++nt) {
    const int n0 = nt * 16;
    const float* krow = k + ((size_t)(b * S_) + n0 + ncol) * H_;
    v16h bk0 = load_pairs(krow,      lane, 1.0f, false);
    v16h bk1 = load_pairs(krow + 32, lane, 1.0f, false);
    v8f s = {};
    s = wmma16(aq0, bk0, s);
    s = wmma16(aq1, bk1, s);
#pragma unroll
    for (int r = 0; r < 8; ++r) {
      const int m = mhalf + r;
      float x = s[r];
      if (mrow[(size_t)m * S_ + n0 + ncol]) x = SENT;
      float mn = fmaxf(rm[r], x);
      rs[r] = rs[r] * __expf(rm[r] - mn) + __expf(x - mn);
      rm[r] = mn;
    }
  }
  // Reduce (max,sum) across the 16 lanes of each half-wave.
#pragma unroll
  for (int r = 0; r < 8; ++r) {
#pragma unroll
    for (int off = 1; off < 16; off <<= 1) {
      float m2 = __shfl_xor(rm[r], off, 16);
      float s2 = __shfl_xor(rs[r], off, 16);
      float mn = fmaxf(rm[r], m2);
      rs[r] = rs[r] * __expf(rm[r] - mn) + s2 * __expf(m2 - mn);
      rm[r] = mn;
    }
  }
  float alphar[8];
#pragma unroll
  for (int r = 0; r < 8; ++r) alphar[r] = rm[r] + __logf(rs[r]);

  // ---- Phase 2: recompute scores, write normalized attn, P*V ----
  v8f acc[4];
#pragma unroll
  for (int hb = 0; hb < 4; ++hb) acc[hb] = {};

  for (int ntp = 0; ntp < S_ / 32; ++ntp) {
    const int kk0 = ntp * 32;
#pragma unroll
    for (int t = 0; t < 2; ++t) {
      const int n0 = kk0 + t * 16;
      const float* krow = k + ((size_t)(b * S_) + n0 + ncol) * H_;
      v16h bk0 = load_pairs(krow,      lane, 1.0f, false);
      v16h bk1 = load_pairs(krow + 32, lane, 1.0f, false);
      v8f s = {};
      s = wmma16(aq0, bk0, s);
      s = wmma16(aq1, bk1, s);
#pragma unroll
      for (int r = 0; r < 8; ++r) {
        const int m = mhalf + r;
        float x = s[r];
        if (mrow[(size_t)m * S_ + n0 + ncol]) x = SENT;
        float p = __expf(x - alphar[r]);                 // normalized prob
        attn[((size_t)(b * S_) + q0 + m) * S_ + n0 + ncol] = p;
        ptile[w][m][t * 16 + ncol] = (_Float16)p;        // stage for A-layout
      }
    }
    asm volatile("s_wait_dscnt 0" ::: "memory");  // DS stores visible to reads
    v16h ap;
#pragma unroll
    for (int j = 0; j < 8; ++j) {
      int kk = kpatA(j, lane);
      ap[2 * j]     = ptile[w][ncol][kk];
      ap[2 * j + 1] = ptile[w][ncol][kk + 1];
    }
#pragma unroll
    for (int hb = 0; hb < 4; ++hb) {
      const float* vb = v + ((size_t)(b * S_) + kk0) * H_ + hb * 16 + ncol;
      v16h bv = load_vtile(vb, lane);
      acc[hb] = wmma16(ap, bv, acc[hb]);
    }
  }

  // ---- ctx writeback (already normalized) ----
#pragma unroll
  for (int hb = 0; hb < 4; ++hb)
#pragma unroll
    for (int r = 0; r < 8; ++r)
      ctx[((size_t)(b * S_) + q0 + mhalf + r) * H_ + hb * 16 + ncol] =
          acc[hb][r];
}

extern "C" void kernel_launch(void* const* d_in, const int* in_sizes, int n_in,
                              void* d_out, int out_size, void* d_ws,
                              size_t ws_size, hipStream_t stream) {
  const float* q = (const float*)d_in[0];
  const float* k = (const float*)d_in[1];
  const float* v = (const float*)d_in[2];
  const unsigned char* mask = (const unsigned char*)d_in[3];  // jax bool = u8
  // d_in[4] = cca scalar; reference path is cca==0.

  float* out  = (float*)d_out;
  float* ctx  = out;                                // [B,S,H] first
  float* attn = out + (size_t)B_ * S_ * H_;         // then [B,S,S]

  dim3 grid(B_ * (S_ / 16) / WAVES_PER_WG);         // 1024 WGs
  dim3 block(WAVES_PER_WG * 32);                    // 4 waves (wave32)
  hipLaunchKernelGGL(sdpa_fwd_kernel, grid, block, 0, stream,
                     q, k, v, mask, ctx, attn);
}
